// CompositionLayer_52707838657224
// MI455X (gfx1250) — matrix operand order repaired and checked by reading, not compiled, not run on hardware
//
#include <hip/hip_runtime.h>
#include <hip/hip_bf16.h>
#include <math.h>

// ---------------- problem constants ----------------
#define BB   8
#define SS   2048
#define HH   768
#define CHN  3072
#define WW   1024
#define MM   (BB * WW)        // 8192 words
#define K2H  (2 * HH)         // 1536
#define EPSL 1e-5f

// GEMM staging geometry: block tile 64 rows x 256 cols, k-step 32 (bf16)
#define LINES 320             // 64 A lines + 256 B lines
#define LINEB 64              // 32 K * 2B per line
#define BUFB  (LINES * LINEB) // 20480 B per buffer (double buffered -> 40 KB)

typedef __attribute__((ext_vector_type(16))) __bf16 v16bf;
typedef __attribute__((ext_vector_type(8)))  __bf16 v8bf;
typedef __attribute__((ext_vector_type(8)))  float  v8f;

__device__ __forceinline__ v16bf bf_cat(v8bf lo, v8bf hi) {
  return __builtin_shufflevector(lo, hi, 0, 1, 2, 3, 4, 5, 6, 7,
                                         8, 9, 10, 11, 12, 13, 14, 15);
}

// ---------------------------------------------------------------------------
// Weight convert: fp32 [K,N] row-major  ->  bf16 [N,K] row-major (transposed)
// ---------------------------------------------------------------------------
__global__ void convT_kernel(const float* __restrict__ src,
                             __bf16* __restrict__ dst, int K, int N) {
  int idx = blockIdx.x * blockDim.x + threadIdx.x;
  if (idx >= K * N) return;
  int k = idx / N;
  int n = idx - k * N;
  dst[(size_t)n * K + k] = (__bf16)src[idx];
}

// ---------------------------------------------------------------------------
// scores[b,s] = x[b,s,:] . w_score + b_score     (one wave32 per row)
// ---------------------------------------------------------------------------
__global__ void score_kernel(const float* __restrict__ x,
                             const float* __restrict__ w_score,
                             const float* __restrict__ b_score,
                             float* __restrict__ scores) {
  int row  = (blockIdx.x * blockDim.x + threadIdx.x) >> 5;   // 0..B*S-1
  int lane = threadIdx.x & 31;
  const float* xr = x + (size_t)row * HH;
  float s = 0.f;
  for (int h = lane; h < HH; h += 32) s += xr[h] * w_score[h];
#pragma unroll
  for (int off = 16; off > 0; off >>= 1) s += __shfl_xor(s, off, 32);
  if (lane == 0) scores[row] = s + b_score[0];
}

// ---------------------------------------------------------------------------
// Per-word span pooling: mean pool + masked-softmax attention pool.
// Writes pooled/att fp32 [M,H] and bf16 concat PA [M,2H].
// ---------------------------------------------------------------------------
__global__ void pool_kernel(const float* __restrict__ x,
                            const int* __restrict__ spans,
                            const float* __restrict__ scores,
                            float* __restrict__ pooled,
                            float* __restrict__ att,
                            __bf16* __restrict__ PA) {
  int m  = blockIdx.x;            // word index, 0..M-1
  int b  = m >> 10;               // W = 1024
  int st = spans[2 * m];
  int en = spans[2 * m + 1];
  bool valid = (st >= 0) && (en > st);
  int L = valid ? (en - st) : 0;
  if (L > 8) L = 8;

  float w[8];
  if (valid) {
    float mx = -1e30f;
#pragma unroll
    for (int i = 0; i < 8; ++i)
      if (i < L) { float sc = scores[b * SS + st + i]; mx = fmaxf(mx, sc); }
    float sum = 0.f;
#pragma unroll
    for (int i = 0; i < 8; ++i) {
      if (i < L) { w[i] = __expf(scores[b * SS + st + i] - mx); sum += w[i]; }
      else w[i] = 0.f;
    }
    float inv = 1.f / sum;
#pragma unroll
    for (int i = 0; i < 8; ++i) w[i] *= inv;
  }

  float invL = valid ? (1.f / (float)L) : 0.f;
  for (int h = threadIdx.x; h < HH; h += blockDim.x) {
    float p = 0.f, a = 0.f;
    if (valid) {
#pragma unroll
      for (int i = 0; i < 8; ++i) {
        if (i < L) {
          float v = x[((size_t)(b * SS + st + i)) * HH + h];
          p += v;
          a += w[i] * v;
        }
      }
    }
    p *= invL;
    size_t o = (size_t)m * HH + h;
    pooled[o] = p;
    att[o]    = a;
    PA[(size_t)m * K2H + h]      = (__bf16)p;
    PA[(size_t)m * K2H + HH + h] = (__bf16)a;
  }
}

// ---------------------------------------------------------------------------
// GEMM core with async-to-LDS double-buffered staging.
//   A  : bf16 [M,K] row-major (lda = K)
//   Bt : bf16 [N,K] row-major (ldb = K), i.e. transposed weights
// Block: 256 thr = 8 waves; wave = (rowg in 0..1) x (colg in 0..3)
// Wave computes 32x64 tile -> acc[2][4]; block tile 64 x 256.
// LDS buffer: lines 0..63 = A rows, 64..319 = B cols, 64B (= 32 K) per line.
// The async target offset is derived from the smem generic pointer: on
// amdgcn a generic LDS address is {SHARED_BASE, lds_offset[31:0]}, so the
// low 32 bits are exactly the VDST byte offset.  The ptrtoint + pointer asm
// operand also force the LDS object to escape, so the "memory" clobber keeps
// the post-barrier ds_load fragment reads alive.
// ---------------------------------------------------------------------------
__device__ __forceinline__ void gemm_core(const __bf16* __restrict__ A, int lda,
                                          const __bf16* __restrict__ Bt, int ldb,
                                          int K, char* smem,
                                          v8f acc[2][4], int& row0, int& n0) {
  const int t      = threadIdx.x;
  const int wave   = t >> 5;
  const int lane   = t & 31;
  const int half   = lane >> 4;
  const int l16    = lane & 15;
  const int colg   = wave & 3;           // 4 col groups * 64
  const int rowg   = wave >> 2;          // 2 row groups * 32
  const int rowblk = blockIdx.y * 64;
  const int nblk   = blockIdx.x * 256;
  row0 = rowblk + rowg * 32;
  n0   = nblk + colg * 64;

  // Per-thread staging chunks: 320 lines * 4 x 16B chunks = 1280, 5 per thread.
  unsigned long long gaddr[5];
  unsigned loff[5];
#pragma unroll
  for (int i = 0; i < 5; ++i) {
    int c     = t + i * 256;
    int line  = c >> 2;
    int chunk = c & 3;
    const __bf16* g = (line < 64)
        ? (A  + (size_t)(rowblk + line) * lda)
        : (Bt + (size_t)(nblk + (line - 64)) * ldb);
    gaddr[i] = (unsigned long long)(uintptr_t)(g + chunk * 8);
    // low 32 bits of the generic LDS address == LDS byte offset (captures smem)
    loff[i]  = (unsigned)(uintptr_t)(smem + line * LINEB + chunk * 16);
  }

  auto issue = [&](int k0, int buf) {
#pragma unroll
    for (int i = 0; i < 5; ++i) {
      unsigned long long ga = gaddr[i] + (unsigned long long)k0 * 2u;
      unsigned lo = loff[i] + (unsigned)(buf * BUFB);
      asm volatile("global_load_async_to_lds_b128 %0, %1, off"
                   :: "v"(lo), "v"(ga), "v"(smem)   // %2: LDS alias anchor
                   : "memory");
    }
  };

  issue(0, 0);
  for (int k0 = 0; k0 < K; k0 += 32) {
    int buf = (k0 >> 5) & 1;
    asm volatile("s_wait_asynccnt 0" ::: "memory");
    __syncthreads();                       // staged buffer visible to all waves
    if (k0 + 32 < K) issue(k0 + 32, buf ^ 1);  // overlap fill of other buffer

    const char* sb = smem + buf * BUFB;
#pragma unroll
    for (int sub = 0; sub < 2; ++sub) {
      // A fragment (16x32): elems 0..7 -> K=8*half+e ; elems 8..15 -> +16
      const char* aln = sb + (size_t)(rowg * 32 + sub * 16 + l16) * LINEB;
      v8bf alo = *(const v8bf*)(aln + 16 * half);
      v8bf ahi = *(const v8bf*)(aln + 32 + 16 * half);
      v16bf af = bf_cat(alo, ahi);
#pragma unroll
      for (int j = 0; j < 4; ++j) {
        // B fragment (32x16): col = n0 + j*16 + l16 ; elems 0..7 -> K=8*half+e ; elems 8..15 -> +16
        const char* bln = sb + (size_t)(64 + colg * 64 + j * 16 + l16) * LINEB;
        v8bf blo = *(const v8bf*)(bln + 16 * half);
        v8bf bhi = *(const v8bf*)(bln + 32 + 16 * half);
        v16bf bfv = bf_cat(blo, bhi);
        acc[sub][j] = __builtin_amdgcn_wmma_f32_16x16x32_bf16(
            false, af, false, bfv, (short)0, acc[sub][j], false, false);
      }
    }
    // No second barrier needed: buffer `buf` is only refilled by issue() after
    // the next iteration's barrier, which all waves reach post-compute.
  }
}

#define GEMM_PROLOG(Aptr, lda_, Bptr, ldb_, Kdim)                              \
  __shared__ __align__(16) char smem[2 * BUFB];                                \
  v8f acc[2][4] = {};                                                          \
  int row0, n0;                                                                \
  gemm_core((Aptr), (lda_), (Bptr), (ldb_), (Kdim), smem, acc, row0, n0);      \
  int lane = threadIdx.x & 31;                                                 \
  int half = lane >> 4;                                                        \
  int l16  = lane & 15;

// ---------------------------------------------------------------------------
// GEMM 1: gate = sigmoid(PA @ w_gate + b_gate); fused = g*att + (1-g)*pooled
// ---------------------------------------------------------------------------
__global__ __launch_bounds__(256)
void gemm_gate_kernel(const __bf16* __restrict__ PA,
                      const __bf16* __restrict__ wgT,
                      const float* __restrict__ b_gate,
                      const float* __restrict__ pooled,
                      const float* __restrict__ att,
                      float* __restrict__ fused,
                      __bf16* __restrict__ fusedb) {
  GEMM_PROLOG(PA, K2H, wgT, K2H, K2H)
#pragma unroll
  for (int sub = 0; sub < 2; ++sub) {
#pragma unroll
    for (int j = 0; j < 4; ++j) {
      int n = n0 + j * 16 + l16;
      float bg = b_gate[n];
#pragma unroll
      for (int r = 0; r < 8; ++r) {
        int m = row0 + sub * 16 + 8 * half + r;
        size_t o = (size_t)m * HH + n;
        float g = 1.f / (1.f + __expf(-(acc[sub][j][r] + bg)));
        float f = g * att[o] + (1.f - g) * pooled[o];
        fused[o]  = f;
        fusedb[o] = (__bf16)f;
      }
    }
  }
}

// ---------------------------------------------------------------------------
// GEMM 2: hid = gelu(fused @ w1 + b1), exact erf GELU, bf16 out
// ---------------------------------------------------------------------------
__global__ __launch_bounds__(256)
void gemm_mlp1_kernel(const __bf16* __restrict__ fusedb,
                      const __bf16* __restrict__ w1T,
                      const float* __restrict__ b1,
                      __bf16* __restrict__ hid) {
  GEMM_PROLOG(fusedb, HH, w1T, HH, HH)
#pragma unroll
  for (int sub = 0; sub < 2; ++sub) {
#pragma unroll
    for (int j = 0; j < 4; ++j) {
      int n = n0 + j * 16 + l16;
      float bb = b1[n];
#pragma unroll
      for (int r = 0; r < 8; ++r) {
        int m = row0 + sub * 16 + 8 * half + r;
        float v = acc[sub][j][r] + bb;
        float ge = 0.5f * v * (1.f + erff(v * 0.70710678118654752f));
        hid[(size_t)m * CHN + n] = (__bf16)ge;
      }
    }
  }
}

// ---------------------------------------------------------------------------
// GEMM 3: y = fused + (hid @ w2 + b2)   (pre-LayerNorm residual)
// ---------------------------------------------------------------------------
__global__ __launch_bounds__(256)
void gemm_mlp2_kernel(const __bf16* __restrict__ hid,
                      const __bf16* __restrict__ w2T,
                      const float* __restrict__ b2,
                      const float* __restrict__ fused,
                      float* __restrict__ y) {
  GEMM_PROLOG(hid, CHN, w2T, CHN, CHN)
#pragma unroll
  for (int sub = 0; sub < 2; ++sub) {
#pragma unroll
    for (int j = 0; j < 4; ++j) {
      int n = n0 + j * 16 + l16;
      float bb = b2[n];
#pragma unroll
      for (int r = 0; r < 8; ++r) {
        int m = row0 + sub * 16 + 8 * half + r;
        size_t o = (size_t)m * HH + n;
        y[o] = acc[sub][j][r] + bb + fused[o];
      }
    }
  }
}

// ---------------------------------------------------------------------------
// LayerNorm + valid masking + valid/index outputs. One block per word row.
// ---------------------------------------------------------------------------
__global__ __launch_bounds__(256)
void ln_kernel(const float* __restrict__ y,
               const int* __restrict__ spans,
               const float* __restrict__ gamma,
               const float* __restrict__ beta,
               float* __restrict__ composed,
               float* __restrict__ validOut,
               float* __restrict__ idxOut) {
  __shared__ float s1[256];
  __shared__ float s2[256];
  int m = blockIdx.x;
  const float* row = y + (size_t)m * HH;
  float s = 0.f, ss = 0.f;
  for (int h = threadIdx.x; h < HH; h += 256) {
    float v = row[h];
    s += v; ss += v * v;
  }
  s1[threadIdx.x] = s;
  s2[threadIdx.x] = ss;
  __syncthreads();
#pragma unroll
  for (int st = 128; st > 0; st >>= 1) {
    if (threadIdx.x < st) {
      s1[threadIdx.x] += s1[threadIdx.x + st];
      s2[threadIdx.x] += s2[threadIdx.x + st];
    }
    __syncthreads();
  }
  float mu  = s1[0] * (1.f / HH);
  float var = s2[0] * (1.f / HH) - mu * mu;
  float rs  = rsqrtf(var + EPSL);
  int start = spans[2 * m], end = spans[2 * m + 1];
  bool valid = (start >= 0) && (end > start);
  for (int h = threadIdx.x; h < HH; h += 256) {
    float o = (row[h] - mu) * rs * gamma[h] + beta[h];
    composed[(size_t)m * HH + h] = valid ? o : 0.f;
  }
  if (threadIdx.x == 0) {
    validOut[m] = valid ? 1.f : 0.f;
    idxOut[m]   = valid ? (float)start : -1.f;
  }
}

// ---------------------------------------------------------------------------
extern "C" void kernel_launch(void* const* d_in, const int* in_sizes, int n_in,
                              void* d_out, int out_size, void* d_ws, size_t ws_size,
                              hipStream_t stream) {
  (void)in_sizes; (void)n_in; (void)out_size; (void)ws_size;
  const float* x        = (const float*)d_in[0];
  const int*   spans    = (const int*)  d_in[1];
  const float* w_score  = (const float*)d_in[2];
  const float* b_score  = (const float*)d_in[3];
  const float* w_gate   = (const float*)d_in[4];
  const float* b_gate   = (const float*)d_in[5];
  const float* w1       = (const float*)d_in[6];
  const float* b1       = (const float*)d_in[7];
  const float* w2       = (const float*)d_in[8];
  const float* b2       = (const float*)d_in[9];
  const float* gamma    = (const float*)d_in[10];
  const float* beta     = (const float*)d_in[11];

  // workspace carve-up (256B aligned slabs)
  char* ws = (char*)d_ws;
  size_t off = 0;
  auto alloc = [&](size_t bytes) {
    char* p = ws + off;
    off += (bytes + 255) & ~(size_t)255;
    return p;
  };
  __bf16* wgT    = (__bf16*)alloc((size_t)HH  * K2H * 2);   // w_gate^T [H ,2H]
  __bf16* w1T    = (__bf16*)alloc((size_t)CHN * HH  * 2);   // w1^T     [CH,H ]
  __bf16* w2T    = (__bf16*)alloc((size_t)HH  * CHN * 2);   // w2^T     [H ,CH]
  float*  scores = (float*) alloc((size_t)BB * SS * 4);
  float*  pooled = (float*) alloc((size_t)MM * HH * 4);
  float*  att    = (float*) alloc((size_t)MM * HH * 4);
  __bf16* PA     = (__bf16*)alloc((size_t)MM * K2H * 2);
  float*  fused  = (float*) alloc((size_t)MM * HH * 4);
  __bf16* fusedb = (__bf16*)alloc((size_t)MM * HH * 2);
  __bf16* hid    = (__bf16*)alloc((size_t)MM * CHN * 2);
  float*  yres   = (float*) alloc((size_t)MM * HH * 4);

  float* composed = (float*)d_out;
  float* validOut = composed + (size_t)MM * HH;
  float* idxOut   = validOut + MM;

  // 1) weights -> transposed bf16
  {
    int n;
    n = K2H * HH;  convT_kernel<<<(n + 255) / 256, 256, 0, stream>>>(w_gate, wgT, K2H, HH);
    n = HH * CHN;  convT_kernel<<<(n + 255) / 256, 256, 0, stream>>>(w1,     w1T, HH,  CHN);
    n = CHN * HH;  convT_kernel<<<(n + 255) / 256, 256, 0, stream>>>(w2,     w2T, CHN, HH);
  }
  // 2) attention scores (wave per row)
  score_kernel<<<(BB * SS * 32) / 256, 256, 0, stream>>>(x, w_score, b_score, scores);
  // 3) span pooling (block per word)
  pool_kernel<<<MM, 256, 0, stream>>>(x, spans, scores, pooled, att, PA);
  // 4) gate GEMM + gated fuse   [8192 x 1536] @ [1536 x 768]
  gemm_gate_kernel<<<dim3(HH / 256, MM / 64), 256, 0, stream>>>(
      PA, wgT, b_gate, pooled, att, fused, fusedb);
  // 5) MLP up + GELU            [8192 x 768] @ [768 x 3072]
  gemm_mlp1_kernel<<<dim3(CHN / 256, MM / 64), 256, 0, stream>>>(
      fusedb, w1T, b1, hid);
  // 6) MLP down + residual      [8192 x 3072] @ [3072 x 768]
  gemm_mlp2_kernel<<<dim3(HH / 256, MM / 64), 256, 0, stream>>>(
      hid, w2T, b2, fused, yres);
  // 7) LayerNorm + masking + valid/index
  ln_kernel<<<MM, 256, 0, stream>>>(yres, spans, gamma, beta,
                                    composed, validOut, idxOut);
}
